// PAdicAttention_52003464020147
// MI455X (gfx1250) — matrix operand order repaired and checked by reading, not compiled, not run on hardware
//
#include <hip/hip_runtime.h>
#include <hip/hip_bf16.h>
#include <math.h>

// ---------------------------------------------------------------------------
// P-adic tree attention for MI455X (gfx1250, wave32, WMMA).
// Pipeline:
//   cvt:  x,Wp,Wv,Wo  f32 -> bf16 (raw u16 storage)
//   gemm<2>: P   = sigmoid(x@Wp + bp)      (WMMA bf16, fused epilogue)
//   gemm<1>: V   = x@Wv  -> bf16           (WMMA bf16)
//   attn:    ctx = softmax(causal sim)@V   (VALU sim + WMMA bf16 PV)
//   gemm<0>: out = ctx@Wo -> f32           (WMMA bf16)
// GEMM is register-blocked: 128x64 block tile, each wave owns 2 A fragments
// so every B fragment gathered from LDS feeds two v_wmma ops.
// ---------------------------------------------------------------------------

typedef __bf16 v16bf __attribute__((ext_vector_type(16)));
typedef float  v8f   __attribute__((ext_vector_type(8)));

#define D_MODEL 1024
#define N_HEADS 16
#define TREE_DEPTH 5
#define BB 2
#define TT 1024
#define DH 64                       // D_MODEL / N_HEADS
#define HP (N_HEADS * TREE_DEPTH)   // 80
#define MT (BB * TT)                // 2048 rows

__device__ __forceinline__ unsigned short f32_to_bf16(float f) {
  union { float f; unsigned int u; } x; x.f = f;
  unsigned int u = x.u;
  unsigned int r = u + 0x7fffu + ((u >> 16) & 1u);   // round-to-nearest-even
  if ((u & 0x7f800000u) == 0x7f800000u) r = u;       // inf/nan passthrough
  return (unsigned short)(r >> 16);
}

__global__ void cvt_f32_bf16(const float* __restrict__ in,
                             unsigned short* __restrict__ out, int n) {
  int i = blockIdx.x * blockDim.x + threadIdx.x;
  int stride = gridDim.x * blockDim.x;
  for (; i < n; i += stride) out[i] = f32_to_bf16(in[i]);
}

// ---------------------------------------------------------------------------
// Tiled WMMA GEMM: C[M,N] = A[M,K] (bf16 row-major) * B[K,N] (bf16 row-major)
// Block = 128x64 tile, 128 threads = 4 waves, each wave owns 32 rows x 64 cols
// (two 16-row A fragments). BK = 32 -> 8 v_wmma_f32_16x16x32_bf16 per K-step
// per wave, with each B fragment reused by two WMMAs.
// EPI: 0 = store f32, 1 = store bf16, 2 = store f32 sigmoid(acc + bias[n]).
// M multiple of 128, K multiple of 32 (true at all call sites); N is guarded.
// ---------------------------------------------------------------------------
template <int EPI>
__global__ __launch_bounds__(128)
void gemm_wmma_bf16(const unsigned short* __restrict__ A,
                    const unsigned short* __restrict__ B,
                    const float* __restrict__ bias,
                    void* __restrict__ Cout,
                    int M, int N, int K) {
  __shared__ unsigned short sA[128 * 34];  // [m][k], stride 34 (pad)
  __shared__ unsigned short sB[64 * 34];   // [n][k], transposed, stride 34

  const int tid  = threadIdx.x;
  const int lane = tid & 31;
  const int wave = tid >> 5;
  const int half = lane >> 4;
  const int l16  = lane & 15;
  const int m0   = blockIdx.y * 128;
  const int n0   = blockIdx.x * 64;

  v8f acc[2][4];
#pragma unroll
  for (int g = 0; g < 2; ++g)
#pragma unroll
    for (int i = 0; i < 4; ++i) acc[g][i] = 0.f;

  for (int kt = 0; kt < K; kt += 32) {
    // Prefetch next K tile into L2/WGP$ (global_prefetch_b8)
    if (kt + 32 < K) {
      __builtin_prefetch(A + (size_t)(m0 + tid) * K + kt + 32, 0, 1);
      if (n0 + 2 * (tid & 31) < N)
        __builtin_prefetch(
            B + (size_t)(kt + 32 + (tid >> 5) * 8) * N + n0 + 2 * (tid & 31),
            0, 1);
    }

    // Stage A tile 128x32 (dword = 2 contiguous k): 16 dwords per thread
#pragma unroll
    for (int i = 0; i < 16; ++i) {
      int idx = tid + i * 128;            // 0..2047 dwords
      int row = idx >> 4;
      int c2  = idx & 15;
      unsigned int u =
          *(const unsigned int*)(A + (size_t)(m0 + row) * K + kt + 2 * c2);
      *(unsigned int*)(sA + row * 34 + 2 * c2) = u;
    }
    // Stage B tile 32x64, transposed to [n][k] so K-pairs are contiguous
#pragma unroll
    for (int i = 0; i < 8; ++i) {
      int idx = tid + i * 128;
      int kk  = idx >> 5;
      int nc  = idx & 31;
      int n   = n0 + 2 * nc;
      unsigned int u = 0;
      if (n < N)
        u = *(const unsigned int*)(B + (size_t)(kt + kk) * N + n);
      sB[(2 * nc) * 34 + kk]     = (unsigned short)(u & 0xffffu);
      sB[(2 * nc + 1) * 34 + kk] = (unsigned short)(u >> 16);
    }
    __syncthreads();

    // Two A fragments per wave: rows 32*wave + {0,16} + l16.
    // 16-bit A 16x32 layout (lanes 0-15: K 0-7,16-23; +8 for hi half).
    union { v16bf v; unsigned int u[8]; } a[2];
#pragma unroll
    for (int g = 0; g < 2; ++g) {
      const unsigned short* ar = sA + (wave * 32 + g * 16 + l16) * 34;
#pragma unroll
      for (int j = 0; j < 8; ++j) {
        int k0 = (j < 4) ? (2 * j + 8 * half) : (16 + 2 * (j - 4) + 8 * half);
        a[g].u[j] = *(const unsigned int*)(ar + k0);
      }
    }
#pragma unroll
    for (int nt = 0; nt < 4; ++nt) {
      union { v16bf v; unsigned int u[8]; } bf;
      const unsigned short* br = sB + (nt * 16 + l16) * 34;
#pragma unroll
      for (int v = 0; v < 8; ++v)
        bf.u[v] = *(const unsigned int*)(br + ((v < 4) ? (2 * v + 8 * half) : (16 + 2 * (v - 4) + 8 * half)));
      acc[0][nt] = __builtin_amdgcn_wmma_f32_16x16x32_bf16(
          false, a[0].v, false, bf.v, (short)0, acc[0][nt], false, false);
      acc[1][nt] = __builtin_amdgcn_wmma_f32_16x16x32_bf16(
          false, a[1].v, false, bf.v, (short)0, acc[1][nt], false, false);
    }
    __syncthreads();
  }

  // Epilogue. C 16x16 f32 layout: VGPR r, lane -> (M = r + 8*half, N = l16)
#pragma unroll
  for (int g = 0; g < 2; ++g) {
#pragma unroll
    for (int nt = 0; nt < 4; ++nt) {
      int n = n0 + nt * 16 + l16;
      if (n >= N) continue;
#pragma unroll
      for (int r = 0; r < 8; ++r) {
        int m = m0 + wave * 32 + g * 16 + r + 8 * half;
        float v = acc[g][nt][r];
        if (EPI == 2) v = 1.0f / (1.0f + __expf(-(v + bias[n])));
        if (EPI == 1)
          ((unsigned short*)Cout)[(size_t)m * N + n] = f32_to_bf16(v);
        else
          ((float*)Cout)[(size_t)m * N + n] = v;
      }
    }
  }
}

// ---------------------------------------------------------------------------
// Fused tree-sim + causal softmax + PV. One wave per (b, h, 16 q rows).
// Pass 1: row maxima over all k (VALU sim, shfl_xor butterfly reduce).
// Pass 2: per 32-k chunk -> exp probs to LDS (bf16), V tile to LDS
//         (transposed [n][k]), 4x v_wmma_f32_16x16x32_bf16 accumulate ctx.
// ---------------------------------------------------------------------------
__global__ __launch_bounds__(32)
void padic_attn(const float* __restrict__ P,
                const unsigned short* __restrict__ V,
                unsigned short* __restrict__ ctx) {
  const int lane = threadIdx.x;
  const int half = lane >> 4;
  const int l16  = lane & 15;
  const int blk  = blockIdx.x;
  const int qt   = blk & 63;
  const int h    = (blk >> 6) & 15;
  const int b    = blk >> 10;
  const int q0   = qt * 16;

  __shared__ float sPq[16][TREE_DEPTH];
  __shared__ unsigned short sAttn[16 * 34];   // [q][k] probs, bf16
  __shared__ unsigned short sV[64 * 34];      // [n][k] V tile, bf16

  if (lane < 16) {
    const float* p = P + (size_t)(b * TT + q0 + lane) * HP + h * TREE_DEPTH;
#pragma unroll
    for (int d = 0; d < TREE_DEPTH; ++d) sPq[lane][d] = p[d];
  }
  __syncthreads();

  const int nch = (q0 + 16 + 31) >> 5;   // chunks of 32 k covering causal range

  // ---- pass 1: row maxima ----
  float mrow[16];
#pragma unroll
  for (int r = 0; r < 16; ++r) mrow[r] = -__builtin_inff();

  for (int c = 0; c < nch; ++c) {
    const int k = c * 32 + lane;
    float pk[TREE_DEPTH];
    const float* pp = P + (size_t)(b * TT + k) * HP + h * TREE_DEPTH;
#pragma unroll
    for (int d = 0; d < TREE_DEPTH; ++d) pk[d] = pp[d];
#pragma unroll
    for (int r = 0; r < 16; ++r) {
      float cum = 1.f, s = 0.f;
#pragma unroll
      for (int d = 0; d < TREE_DEPTH; ++d) {
        float pq = sPq[r][d];
        float ag = pq * pk[d] + (1.f - pq) * (1.f - pk[d]);
        cum *= ag;
        s += cum;
      }
      float sim = s * (1.0f / TREE_DEPTH);
      if (k > q0 + r) sim = -__builtin_inff();
      mrow[r] = fmaxf(mrow[r], sim);
    }
  }
#pragma unroll
  for (int r = 0; r < 16; ++r) {
    float v = mrow[r];
#pragma unroll
    for (int off = 16; off >= 1; off >>= 1)
      v = fmaxf(v, __shfl_xor(v, off, 32));
    mrow[r] = v;
  }

  // ---- pass 2: exp + WMMA accumulate ----
  v8f acc[4];
#pragma unroll
  for (int i = 0; i < 4; ++i) acc[i] = 0.f;
  float lrow[16];
#pragma unroll
  for (int r = 0; r < 16; ++r) lrow[r] = 0.f;

  for (int c = 0; c < nch; ++c) {
    const int kc = c * 32;
    const int k  = kc + lane;

    // V tile (32k x 64n) -> LDS transposed [n][k]; dword = 2 contiguous n
#pragma unroll
    for (int i = 0; i < 32; ++i) {
      unsigned int u = *(const unsigned int*)(
          V + (size_t)(b * TT + kc + i) * D_MODEL + h * DH + 2 * lane);
      sV[(2 * lane) * 34 + i]     = (unsigned short)(u & 0xffffu);
      sV[(2 * lane + 1) * 34 + i] = (unsigned short)(u >> 16);
    }

    float pk[TREE_DEPTH];
    const float* pp = P + (size_t)(b * TT + k) * HP + h * TREE_DEPTH;
#pragma unroll
    for (int d = 0; d < TREE_DEPTH; ++d) pk[d] = pp[d];
#pragma unroll
    for (int r = 0; r < 16; ++r) {
      float cum = 1.f, s = 0.f;
#pragma unroll
      for (int d = 0; d < TREE_DEPTH; ++d) {
        float pq = sPq[r][d];
        float ag = pq * pk[d] + (1.f - pq) * (1.f - pk[d]);
        cum *= ag;
        s += cum;
      }
      float sim = s * (1.0f / TREE_DEPTH);
      if (k > q0 + r) sim = -__builtin_inff();
      float p = __expf(sim - mrow[r]);   // masked -> exp(-inf) = 0
      lrow[r] += p;
      sAttn[r * 34 + lane] = f32_to_bf16(p);
    }
    __syncthreads();

    union { v16bf v; unsigned int u[8]; } a;
    const unsigned short* ar = sAttn + l16 * 34;
#pragma unroll
    for (int j = 0; j < 8; ++j) {
      int k0 = (j < 4) ? (2 * j + 8 * half) : (16 + 2 * (j - 4) + 8 * half);
      a.u[j] = *(const unsigned int*)(ar + k0);
    }
#pragma unroll
    for (int nt = 0; nt < 4; ++nt) {
      union { v16bf v; unsigned int u[8]; } bf;
      const unsigned short* br = sV + (nt * 16 + l16) * 34;
#pragma unroll
      for (int v = 0; v < 8; ++v)
        bf.u[v] = *(const unsigned int*)(br + ((v < 4) ? (2 * v + 8 * half) : (16 + 2 * (v - 4) + 8 * half)));
      acc[nt] = __builtin_amdgcn_wmma_f32_16x16x32_bf16(
          false, a.v, false, bf.v, (short)0, acc[nt], false, false);
    }
    __syncthreads();
  }

  // row-sum reduce, normalize, store ctx (bf16)
#pragma unroll
  for (int r = 0; r < 16; ++r) {
    float v = lrow[r];
#pragma unroll
    for (int off = 16; off >= 1; off >>= 1) v += __shfl_xor(v, off, 32);
    lrow[r] = v;
  }
  float inv[8];
#pragma unroll
  for (int r = 0; r < 8; ++r)
    inv[r] = 1.0f / ((half == 0) ? lrow[r] : lrow[r + 8]);

#pragma unroll
  for (int nt = 0; nt < 4; ++nt) {
    int n = h * DH + nt * 16 + l16;
#pragma unroll
    for (int r = 0; r < 8; ++r) {
      int q = q0 + r + 8 * half;
      ctx[(size_t)(b * TT + q) * D_MODEL + n] = f32_to_bf16(acc[nt][r] * inv[r]);
    }
  }
}

// ---------------------------------------------------------------------------
extern "C" void kernel_launch(void* const* d_in, const int* in_sizes, int n_in,
                              void* d_out, int out_size, void* d_ws,
                              size_t ws_size, hipStream_t stream) {
  const float* x  = (const float*)d_in[0];
  const float* Wp = (const float*)d_in[1];
  const float* bp = (const float*)d_in[2];
  const float* Wv = (const float*)d_in[3];
  const float* Wo = (const float*)d_in[4];
  float* out = (float*)d_out;

  char* ws = (char*)d_ws;
  auto carve = [&](size_t bytes) {
    char* p = ws;
    ws += (bytes + 255) & ~(size_t)255;
    return p;
  };
  unsigned short* xb   = (unsigned short*)carve((size_t)MT * D_MODEL * 2);
  unsigned short* Wvb  = (unsigned short*)carve((size_t)D_MODEL * D_MODEL * 2);
  unsigned short* Wob  = (unsigned short*)carve((size_t)D_MODEL * D_MODEL * 2);
  unsigned short* Wpb  = (unsigned short*)carve((size_t)D_MODEL * HP * 2);
  unsigned short* Vb   = (unsigned short*)carve((size_t)MT * D_MODEL * 2);
  unsigned short* ctxb = (unsigned short*)carve((size_t)MT * D_MODEL * 2);
  float*          Pbuf = (float*)carve((size_t)MT * HP * 4);

  // f32 -> bf16 conversions
  {
    int n;
    n = MT * D_MODEL;
    cvt_f32_bf16<<<(n + 1023) / 1024, 256, 0, stream>>>(x, xb, n);
    n = D_MODEL * D_MODEL;
    cvt_f32_bf16<<<(n + 1023) / 1024, 256, 0, stream>>>(Wv, Wvb, n);
    cvt_f32_bf16<<<(n + 1023) / 1024, 256, 0, stream>>>(Wo, Wob, n);
    n = D_MODEL * HP;
    cvt_f32_bf16<<<(n + 1023) / 1024, 256, 0, stream>>>(Wp, Wpb, n);
  }

  // P = sigmoid(x@Wp + bp)   (2048 x 80 x 1024)
  {
    dim3 g((HP + 63) / 64, MT / 128);
    gemm_wmma_bf16<2><<<g, 128, 0, stream>>>(xb, Wpb, bp, (void*)Pbuf,
                                             MT, HP, D_MODEL);
  }
  // V = x@Wv -> bf16         (2048 x 1024 x 1024)
  {
    dim3 g(D_MODEL / 64, MT / 128);
    gemm_wmma_bf16<1><<<g, 128, 0, stream>>>(xb, Wvb, nullptr, (void*)Vb,
                                             MT, D_MODEL, D_MODEL);
  }
  // ctx = softmax(sim)@V     (one wave per (b,h,16 q rows))
  padic_attn<<<BB * N_HEADS * (TT / 16), 32, 0, stream>>>(Pbuf, Vb, ctxb);

  // out = ctx@Wo -> f32      (2048 x 1024 x 1024)
  {
    dim3 g(D_MODEL / 64, MT / 128);
    gemm_wmma_bf16<0><<<g, 128, 0, stream>>>(ctxb, Wob, nullptr, (void*)out,
                                             MT, D_MODEL, D_MODEL);
  }
}
